// EGNNNet_70789650973261
// MI455X (gfx1250) — compile-verified, matrix operand order checked by reading, not run on hardware
//
#include <hip/hip_runtime.h>

// ---------------------------------------------------------------------------
// CDNA5 / gfx1250 EGNN forward.  All GEMMs run on v_wmma_f32_16x16x32_bf16
// (bf16 A/B, f32 accumulate).  wave32 everywhere.
// ---------------------------------------------------------------------------

typedef __bf16 bf16_t;
typedef __attribute__((ext_vector_type(16))) __bf16 v16bf;
typedef __attribute__((ext_vector_type(8)))  __bf16 v8bf;
typedef __attribute__((ext_vector_type(8)))  float  v8f;

#define ACT_NONE 0
#define ACT_SILU 1

__device__ __forceinline__ v16bf load_frag16(const bf16_t* p) {
  // elements 0..7 from p[0..7], 8..15 from p[16..23]  (ISA 16-bit A/B layout:
  // lane half selects K offset 0/8; per-VGPR pairs cover K and K+16 halves)
  v8bf lo = *(const v8bf*)p;
  v8bf hi = *(const v8bf*)(p + 16);
  return __builtin_shufflevector(lo, hi, 0,1,2,3,4,5,6,7,8,9,10,11,12,13,14,15);
}

// Generic GEMM: C[M,128] = act(A[M,K] * B[K,128] + bias) (+ addC residual)
// A: bf16 row-major, lda >= K (K multiple of 32).  BT: bf16 [128 x ldb]
// (weights stored transposed so fragment loads are contiguous).
// One block = 16 rows of C; 4 waves, each wave owns two 16x16 N-tiles.
__global__ void k_gemm(const bf16_t* __restrict__ A, int lda,
                       const bf16_t* __restrict__ BT, int ldb,
                       const float* __restrict__ bias,
                       const float* __restrict__ addC,
                       float* __restrict__ Cf, bf16_t* __restrict__ Cb,
                       int K, int act)
{
  const int m0   = blockIdx.x * 16;
  const int wave = threadIdx.x >> 5;
  const int lane = threadIdx.x & 31;
  const int hf   = lane >> 4;          // lane half selects K sub-offset
  const int mr   = lane & 15;          // row (A) / col (B) within tile
  const int n0   = wave * 32;          // this wave's two N-tiles

  v8f c0 = {};
  v8f c1 = {};

  const bf16_t* arow = A  + (size_t)(m0 + mr)      * lda + hf * 8;
  const bf16_t* b0r  = BT + (size_t)(n0 + mr)      * ldb + hf * 8;
  const bf16_t* b1r  = BT + (size_t)(n0 + 16 + mr) * ldb + hf * 8;

  for (int k = 0; k < K; k += 32) {
    v16bf af = load_frag16(arow + k);
    v16bf bf0 = load_frag16(b0r + k);
    v16bf bf1 = load_frag16(b1r + k);
    c0 = __builtin_amdgcn_wmma_f32_16x16x32_bf16(false, af, false, bf0,
                                                 (short)0, c0, false, false);
    c1 = __builtin_amdgcn_wmma_f32_16x16x32_bf16(false, af, false, bf1,
                                                 (short)0, c1, false, false);
  }

  // C/D layout: VGPR r -> lanes 0-15: (M=r, N=lane); lanes 16-31: (M=8+r, N=lane-16)
  const int col0 = n0 + (lane & 15);
#pragma unroll
  for (int r = 0; r < 8; ++r) {
    const int row = m0 + r + hf * 8;
#pragma unroll
    for (int t = 0; t < 2; ++t) {
      float v = t ? c1[r] : c0[r];
      const int col = col0 + t * 16;
      if (bias) v += bias[col];
      const size_t idx = (size_t)row * 128 + col;
      if (addC) v += addC[idx];
      if (act == ACT_SILU) v = v / (1.f + __expf(-v));
      if (Cf) Cf[idx] = v;
      if (Cb) Cb[idx] = (bf16_t)v;
    }
  }
}

// Convert weight W[K x N(=128)] f32 -> WT[128 x Kp] bf16 (transposed, zero-padded)
__global__ void k_wt(const float* __restrict__ W, bf16_t* __restrict__ WT,
                     int K, int N, int Kp)
{
  int i = blockIdx.x * 256 + threadIdx.x;
  if (i >= N * Kp) return;
  int n = i / Kp, k = i % Kp;
  WT[i] = (k < K) ? (bf16_t)W[(size_t)k * N + n] : (bf16_t)0.f;
}

// Convert activation matrix f32 [M x K] -> bf16 [M x Kp] zero-padded
__global__ void k_pad_bf16(const float* __restrict__ src, bf16_t* __restrict__ dst,
                           int M, int K, int Kp)
{
  size_t i = (size_t)blockIdx.x * 256 + threadIdx.x;
  if (i >= (size_t)M * Kp) return;
  int r = (int)(i / Kp), c = (int)(i % Kp);
  dst[i] = (c < K) ? (bf16_t)src[(size_t)r * K + c] : (bf16_t)0.f;
}

// LayerNorm in place (+ save a copy). One wave per 128-wide row.
__global__ void k_ln(float* __restrict__ h, const float* __restrict__ g,
                     const float* __restrict__ b, float* __restrict__ hcopy, int N)
{
  int row = blockIdx.x * 4 + (threadIdx.x >> 5);
  int lane = threadIdx.x & 31;
  if (row >= N) return;
  float* hr = h + (size_t)row * 128;
  float v[4], s = 0.f, s2 = 0.f;
#pragma unroll
  for (int j = 0; j < 4; ++j) { v[j] = hr[lane + 32 * j]; s += v[j]; s2 += v[j] * v[j]; }
#pragma unroll
  for (int o = 16; o; o >>= 1) { s += __shfl_xor(s, o, 32); s2 += __shfl_xor(s2, o, 32); }
  float mean = s * (1.f / 128.f);
  float var  = s2 * (1.f / 128.f) - mean * mean;
  float inv  = rsqrtf(var + 1e-5f);
#pragma unroll
  for (int j = 0; j < 4; ++j) {
    int c = lane + 32 * j;
    float x = (v[j] - mean) * inv * g[c] + b[c];
    hr[c] = x;
    if (hcopy) hcopy[(size_t)row * 128 + c] = x;
  }
}

// hn = bf16(elu(LN(h; g,b))).  One wave per row.
__global__ void k_node_prep(const float* __restrict__ h, const float* __restrict__ g,
                            const float* __restrict__ b, bf16_t* __restrict__ hn, int N)
{
  int row = blockIdx.x * 4 + (threadIdx.x >> 5);
  int lane = threadIdx.x & 31;
  if (row >= N) return;
  const float* hr = h + (size_t)row * 128;
  float v[4], s = 0.f, s2 = 0.f;
#pragma unroll
  for (int j = 0; j < 4; ++j) { v[j] = hr[lane + 32 * j]; s += v[j]; s2 += v[j] * v[j]; }
#pragma unroll
  for (int o = 16; o; o >>= 1) { s += __shfl_xor(s, o, 32); s2 += __shfl_xor(s2, o, 32); }
  float mean = s * (1.f / 128.f);
  float var  = s2 * (1.f / 128.f) - mean * mean;
  float inv  = rsqrtf(var + 1e-5f);
#pragma unroll
  for (int j = 0; j < 4; ++j) {
    int c = lane + 32 * j;
    float x = (v[j] - mean) * inv * g[c] + b[c];
    x = x > 0.f ? x : (__expf(x) - 1.f);            // ELU
    hn[(size_t)row * 128 + c] = (bf16_t)x;
  }
}

// Build edge-MLP input row f = [hn[src](128), hn[dst](128), r(1), e(128), pad(31)]
// One block (128 threads) per edge; row stride 416 bf16.
__global__ void k_build_f(const bf16_t* __restrict__ hn, const bf16_t* __restrict__ efe,
                          const float* __restrict__ X,
                          const int* __restrict__ src, const int* __restrict__ dst,
                          bf16_t* __restrict__ F)
{
  int e = blockIdx.x;
  int t = threadIdx.x;
  int s = src[e], d = dst[e];
  bf16_t* row = F + (size_t)e * 416;
  row[t]        = hn[(size_t)s * 128 + t];
  row[128 + t]  = hn[(size_t)d * 128 + t];
  row[257 + t]  = efe[(size_t)e * 128 + t];
  if (t == 0) {
    float dx = X[s*3+0] - X[d*3+0];
    float dy = X[s*3+1] - X[d*3+1];
    float dz = X[s*3+2] - X[d*3+2];
    row[256] = (bf16_t)(dx*dx + dy*dy + dz*dz);
  }
  if (t < 31) row[385 + t] = (bf16_t)0.f;
}

// Per-edge: w = <silu-activated m2, cW2>; scatter w*xhat, count, and h_agg += m.
// One wave per edge.
__global__ void k_edge_reduce(const bf16_t* __restrict__ m2, const bf16_t* __restrict__ m,
                              const float* __restrict__ cW2, const float* __restrict__ X,
                              const int* __restrict__ src, const int* __restrict__ dst,
                              float* __restrict__ x_acc, float* __restrict__ cnt,
                              float* __restrict__ h_agg, int E)
{
  int e = blockIdx.x * 4 + (threadIdx.x >> 5);
  int lane = threadIdx.x & 31;
  if (e >= E) return;
  const bf16_t* m2r = m2 + (size_t)e * 128;
  float w = 0.f;
#pragma unroll
  for (int j = 0; j < 4; ++j) { int c = lane + 32 * j; w += (float)m2r[c] * cW2[c]; }
#pragma unroll
  for (int o = 16; o; o >>= 1) w += __shfl_xor(w, o, 32);
  int s = src[e], d = dst[e];
  const bf16_t* mr = m + (size_t)e * 128;
#pragma unroll
  for (int j = 0; j < 4; ++j) {
    int c = lane + 32 * j;
    atomicAdd(&h_agg[(size_t)d * 128 + c], (float)mr[c]);
  }
  if (lane == 0) {
    float dx = X[s*3+0] - X[d*3+0];
    float dy = X[s*3+1] - X[d*3+1];
    float dz = X[s*3+2] - X[d*3+2];
    float r2 = dx*dx + dy*dy + dz*dz;
    float inv = 1.f / (sqrtf(r2) + 1e-30f);
    atomicAdd(&x_acc[d*3+0], w * dx * inv);
    atomicAdd(&x_acc[d*3+1], w * dy * inv);
    atomicAdd(&x_acc[d*3+2], w * dz * inv);
    atomicAdd(&cnt[d], 1.f);
  }
}

__global__ void k_coord_update(float* __restrict__ X, const float* __restrict__ x_acc,
                               const float* __restrict__ cnt, int N)
{
  int i = blockIdx.x * 256 + threadIdx.x;
  if (i >= N * 3) return;
  X[i] += x_acc[i] / fmaxf(cnt[i / 3], 1.f);
}

__global__ void k_build_nodeA(const bf16_t* __restrict__ hn, const float* __restrict__ h_agg,
                              bf16_t* __restrict__ A, int N)
{
  int i = blockIdx.x * 256 + threadIdx.x;
  if (i >= N * 256) return;
  int r = i >> 8, c = i & 255;
  A[i] = (c < 128) ? hn[(size_t)r * 128 + c]
                   : (bf16_t)h_agg[(size_t)r * 128 + (c - 128)];
}

__global__ void k_merge(const float* __restrict__ hp, const float* __restrict__ hl,
                        float* __restrict__ hc)
{
  size_t i = (size_t)blockIdx.x * 256 + threadIdx.x;
  if (i >= (size_t)3584 * 128) return;
  int row = (int)(i >> 7), c = (int)(i & 127);
  int b = row / 448, n = row % 448;
  hc[i] = (n < 384) ? hp[((size_t)b * 384 + n) * 128 + c]
                    : hl[((size_t)b * 64 + (n - 384)) * 128 + c];
}

__global__ void k_split_res(const float* __restrict__ hc, const float* __restrict__ hp_i,
                            const float* __restrict__ hl_i,
                            float* __restrict__ hp, float* __restrict__ hl)
{
  size_t i = (size_t)blockIdx.x * 256 + threadIdx.x;
  if (i >= (size_t)3584 * 128) return;
  int row = (int)(i >> 7), c = (int)(i & 127);
  int b = row / 448, n = row % 448;
  if (n < 384) {
    size_t j = ((size_t)b * 384 + n) * 128 + c;
    hp[j] = hc[i] + hp_i[j];
  } else {
    size_t j = ((size_t)b * 64 + (n - 384)) * 128 + c;
    hl[j] = hc[i] + hl_i[j];
  }
}

__global__ void k_zero(float* __restrict__ p, int n)
{ int i = blockIdx.x * 256 + threadIdx.x; if (i < n) p[i] = 0.f; }

__global__ void k_copy(const float* __restrict__ s, float* __restrict__ d, int n)
{ int i = blockIdx.x * 256 + threadIdx.x; if (i < n) d[i] = s[i]; }

// ---------------------------------------------------------------------------
// Host orchestration
// ---------------------------------------------------------------------------
namespace {

struct BP {  // block params (sorted-leaf order within the pytree)
  const float *cW1,*cW2,*cb1,*eW1,*eW2,*eb1,*eb2,*ln_b,*ln_g,*nW1,*nW2,*nb1,*nb2;
};
static BP getBP(void* const* d_in, int base) {
  BP b;
  b.cW1 =(const float*)d_in[base+0];  b.cW2 =(const float*)d_in[base+1];
  b.cb1 =(const float*)d_in[base+2];  b.eW1 =(const float*)d_in[base+3];
  b.eW2 =(const float*)d_in[base+4];  b.eb1 =(const float*)d_in[base+5];
  b.eb2 =(const float*)d_in[base+6];  b.ln_b=(const float*)d_in[base+7];
  b.ln_g=(const float*)d_in[base+8];  b.nW1 =(const float*)d_in[base+9];
  b.nW2 =(const float*)d_in[base+10]; b.nb1 =(const float*)d_in[base+11];
  b.nb2 =(const float*)d_in[base+12];
  return b;
}

struct LW { bf16_t *eW1T, *eW2T, *cW1T, *nW1T, *nW2T; };

static inline size_t alignup(size_t x) { return (x + 255) & ~(size_t)255; }

} // namespace

extern "C" void kernel_launch(void* const* d_in, const int* in_sizes, int n_in,
                              void* d_out, int out_size, void* d_ws, size_t ws_size,
                              hipStream_t stream)
{
  (void)in_sizes; (void)n_in; (void)out_size; (void)ws_size;
  const int NPt = 3072, NLt = 512, NCt = 3584;
  const int EP = 73728, EL = 4096, EC = 114688;

  // ---- input pointers (pytree leaf order: sorted dict keys, 'W' < lowercase) ----
  const float* gp_nfeat = (const float*)d_in[0];
  const float* gp_efeat = (const float*)d_in[1];
  const float* gp_coord = (const float*)d_in[2];
  const float* gl_nfeat = (const float*)d_in[3];
  const float* gl_efeat = (const float*)d_in[4];
  const float* gl_coord = (const float*)d_in[5];
  const float* gc_efeat = (const float*)d_in[6];
  const float* gc_coord = (const float*)d_in[7];
  const float* Wce = (const float*)d_in[8];
  const float* Wl  = (const float*)d_in[9];
  const float* Wle = (const float*)d_in[10];
  const float* Wp  = (const float*)d_in[11];
  const float* Wpe = (const float*)d_in[12];
  const float* bce = (const float*)d_in[13];
  const float* bl  = (const float*)d_in[14];
  const float* ble = (const float*)d_in[15];
  const float* bp  = (const float*)d_in[16];
  const float* bpe = (const float*)d_in[17];
  BP bpComp = getBP(d_in, 18);
  BP bpLig  = getBP(d_in, 31);
  const float* lnorm_b = (const float*)d_in[44];
  const float* lnorm_g = (const float*)d_in[45];
  const float* pnorm_b = (const float*)d_in[46];
  const float* pnorm_g = (const float*)d_in[47];
  BP bpProt = getBP(d_in, 48);
  const int* gp_src = (const int*)d_in[61];
  const int* gp_dst = (const int*)d_in[62];
  const int* gl_src = (const int*)d_in[63];
  const int* gl_dst = (const int*)d_in[64];
  const int* gc_src = (const int*)d_in[65];
  const int* gc_dst = (const int*)d_in[66];

  // ---- workspace bump allocator ----
  char* w = (char*)d_ws;
  size_t off = 0;
  auto alloc = [&](size_t bytes) -> void* {
    void* p = w + off; off = alignup(off + bytes); return p;
  };

  BP bps[3] = { bpProt, bpLig, bpComp };
  LW lw[3][3];
  for (int g = 0; g < 3; ++g)
    for (int l = 0; l < 3; ++l) {
      lw[g][l].eW1T = (bf16_t*)alloc(128 * 416 * 2);
      lw[g][l].eW2T = (bf16_t*)alloc(128 * 128 * 2);
      lw[g][l].cW1T = (bf16_t*)alloc(128 * 128 * 2);
      lw[g][l].nW1T = (bf16_t*)alloc(128 * 256 * 2);
      lw[g][l].nW2T = (bf16_t*)alloc(128 * 128 * 2);
    }
  bf16_t* WpT  = (bf16_t*)alloc(128 * 64 * 2);
  bf16_t* WlT  = (bf16_t*)alloc(128 * 64 * 2);
  bf16_t* WpeT = (bf16_t*)alloc(128 * 32 * 2);
  bf16_t* WleT = (bf16_t*)alloc(128 * 32 * 2);
  bf16_t* WceT = (bf16_t*)alloc(128 * 32 * 2);

  float* hp   = (float*)alloc((size_t)NPt * 128 * 4);
  float* hl   = (float*)alloc((size_t)NLt * 128 * 4);
  float* hc   = (float*)alloc((size_t)NCt * 128 * 4);
  float* hp_i = (float*)alloc((size_t)NPt * 128 * 4);
  float* hl_i = (float*)alloc((size_t)NLt * 128 * 4);
  float* cp   = (float*)alloc((size_t)NPt * 3 * 4);
  float* cl   = (float*)alloc((size_t)NLt * 3 * 4);
  float* cc   = (float*)alloc((size_t)NCt * 3 * 4);
  bf16_t* epb = (bf16_t*)alloc((size_t)EP * 128 * 2);
  bf16_t* elb = (bf16_t*)alloc((size_t)EL * 128 * 2);
  bf16_t* ecb = (bf16_t*)alloc((size_t)EC * 128 * 2);
  bf16_t* hn  = (bf16_t*)alloc((size_t)NCt * 128 * 2);
  bf16_t* F   = (bf16_t*)alloc((size_t)EC * 416 * 2);   // also reused as pad scratch
  bf16_t* mA  = (bf16_t*)alloc((size_t)EC * 128 * 2);
  bf16_t* mB  = (bf16_t*)alloc((size_t)EC * 128 * 2);
  float* x_acc = (float*)alloc((size_t)NCt * 3 * 4);
  float* cnt   = (float*)alloc((size_t)NCt * 4);
  float* h_agg = (float*)alloc((size_t)NCt * 128 * 4);
  bf16_t* nodeA = (bf16_t*)alloc((size_t)NCt * 256 * 2);
  bf16_t* tn    = (bf16_t*)alloc((size_t)NCt * 128 * 2);

  auto gemm = [&](const bf16_t* A, int lda, const bf16_t* BT, int ldb,
                  const float* bias, const float* addC, float* Cf, bf16_t* Cb,
                  int M, int K, int act) {
    k_gemm<<<M / 16, 128, 0, stream>>>(A, lda, BT, ldb, bias, addC, Cf, Cb, K, act);
  };
  auto grid1 = [](int n) { return (unsigned)((n + 255) / 256); };

  // ---- 1. weight conversion (transposed, padded bf16) ----
  for (int g = 0; g < 3; ++g) {
    const BP& b = bps[g];
    for (int l = 0; l < 3; ++l) {
      k_wt<<<grid1(128*416),256,0,stream>>>(b.eW1 + (size_t)l*385*128, lw[g][l].eW1T, 385, 128, 416);
      k_wt<<<grid1(128*128),256,0,stream>>>(b.eW2 + (size_t)l*128*128, lw[g][l].eW2T, 128, 128, 128);
      k_wt<<<grid1(128*128),256,0,stream>>>(b.cW1 + (size_t)l*128*128, lw[g][l].cW1T, 128, 128, 128);
      k_wt<<<grid1(128*256),256,0,stream>>>(b.nW1 + (size_t)l*256*128, lw[g][l].nW1T, 256, 128, 256);
      k_wt<<<grid1(128*128),256,0,stream>>>(b.nW2 + (size_t)l*128*128, lw[g][l].nW2T, 128, 128, 128);
    }
  }
  k_wt<<<grid1(128*64),256,0,stream>>>(Wp,  WpT,  62, 128, 64);
  k_wt<<<grid1(128*64),256,0,stream>>>(Wl,  WlT,  62, 128, 64);
  k_wt<<<grid1(128*32),256,0,stream>>>(Wpe, WpeT, 12, 128, 32);
  k_wt<<<grid1(128*32),256,0,stream>>>(Wle, WleT, 12, 128, 32);
  k_wt<<<grid1(128*32),256,0,stream>>>(Wce, WceT, 14, 128, 32);

  // ---- 2. input projections + LN (F reused as pad scratch) ----
  k_pad_bf16<<<grid1(NPt*64),256,0,stream>>>(gp_nfeat, F, NPt, 62, 64);
  gemm(F, 64, WpT, 64, bp, nullptr, hp, nullptr, NPt, 64, ACT_NONE);
  k_ln<<<(NPt+3)/4,128,0,stream>>>(hp, pnorm_g, pnorm_b, hp_i, NPt);

  k_pad_bf16<<<grid1(NLt*64),256,0,stream>>>(gl_nfeat, F, NLt, 62, 64);
  gemm(F, 64, WlT, 64, bl, nullptr, hl, nullptr, NLt, 64, ACT_NONE);
  k_ln<<<(NLt+3)/4,128,0,stream>>>(hl, lnorm_g, lnorm_b, hl_i, NLt);

  k_pad_bf16<<<grid1(EP*32),256,0,stream>>>(gp_efeat, F, EP, 12, 32);
  gemm(F, 32, WpeT, 32, bpe, nullptr, nullptr, epb, EP, 32, ACT_NONE);
  k_pad_bf16<<<grid1(EL*32),256,0,stream>>>(gl_efeat, F, EL, 12, 32);
  gemm(F, 32, WleT, 32, ble, nullptr, nullptr, elb, EL, 32, ACT_NONE);
  k_pad_bf16<<<grid1(EC*32),256,0,stream>>>(gc_efeat, F, EC, 14, 32);
  gemm(F, 32, WceT, 32, bce, nullptr, nullptr, ecb, EC, 32, ACT_NONE);

  hipMemcpyAsync(cp, gp_coord, (size_t)NPt * 3 * 4, hipMemcpyDeviceToDevice, stream);
  hipMemcpyAsync(cl, gl_coord, (size_t)NLt * 3 * 4, hipMemcpyDeviceToDevice, stream);
  hipMemcpyAsync(cc, gc_coord, (size_t)NCt * 3 * 4, hipMemcpyDeviceToDevice, stream);

  // ---- 3. EGNN layer helper ----
  struct GD { int N, E; const int *src, *dst; float *h, *X; bf16_t* eb; int blk; };
  GD gd[3] = {
    { NPt, EP, gp_src, gp_dst, hp, cp, epb, 0 },
    { NLt, EL, gl_src, gl_dst, hl, cl, elb, 1 },
    { NCt, EC, gc_src, gc_dst, hc, cc, ecb, 2 },
  };

  auto egnn = [&](GD& G, int l) {
    const BP& b = bps[G.blk];
    const LW& L = lw[G.blk][l];
    // pre-norm + ELU -> hn (bf16)
    k_node_prep<<<(G.N+3)/4,128,0,stream>>>(G.h, b.ln_g + l*128, b.ln_b + l*128, hn, G.N);
    // edge concat f
    k_build_f<<<G.E,128,0,stream>>>(hn, G.eb, G.X, G.src, G.dst, F);
    // edge MLP (two SiLU layers) -> m in mB
    gemm(F,  416, L.eW1T, 416, b.eb1 + l*128, nullptr, nullptr, mA, G.E, 416, ACT_SILU);
    gemm(mA, 128, L.eW2T, 128, b.eb2 + l*128, nullptr, nullptr, mB, G.E, 128, ACT_SILU);
    // coord branch first layer -> m2 in mA
    gemm(mB, 128, L.cW1T, 128, b.cb1 + l*128, nullptr, nullptr, mA, G.E, 128, ACT_SILU);
    // aggregation
    k_zero<<<grid1(G.N*3),256,0,stream>>>(x_acc, G.N*3);
    k_zero<<<grid1(G.N),256,0,stream>>>(cnt, G.N);
    k_zero<<<grid1(G.N*128),256,0,stream>>>(h_agg, G.N*128);
    k_edge_reduce<<<G.E/4,128,0,stream>>>(mA, mB, b.cW2 + l*128, G.X,
                                          G.src, G.dst, x_acc, cnt, h_agg, G.E);
    k_coord_update<<<grid1(G.N*3),256,0,stream>>>(G.X, x_acc, cnt, G.N);
    // node MLP: silu([hn, h_agg] @ nW1 + nb1) @ nW2 + nb2 + h  (residual fused)
    k_build_nodeA<<<grid1(G.N*256),256,0,stream>>>(hn, h_agg, nodeA, G.N);
    gemm(nodeA, 256, L.nW1T, 256, b.nb1 + l*128, nullptr, nullptr, tn, G.N, 256, ACT_SILU);
    gemm(tn, 128, L.nW2T, 128, b.nb2 + l*128, G.h, G.h, nullptr, G.N, 128, ACT_NONE);
  };

  // ---- 4. layers ----
  for (int l = 0; l < 3; ++l) {
    egnn(gd[0], l);                                   // protein
    egnn(gd[1], l);                                   // ligand
    k_merge<<<grid1(NCt*128),256,0,stream>>>(hp, hl, hc);
    egnn(gd[2], l);                                   // complex
    k_split_res<<<grid1(NCt*128),256,0,stream>>>(hc, hp_i, hl_i, hp, hl);
  }

  // ---- 5. outputs: concat(hp, hl, hc) ----
  float* out = (float*)d_out;
  k_copy<<<grid1(NPt*128),256,0,stream>>>(hp, out, NPt*128);
  k_copy<<<grid1(NLt*128),256,0,stream>>>(hl, out + (size_t)NPt*128, NLt*128);
  k_copy<<<grid1(NCt*128),256,0,stream>>>(hc, out + (size_t)(NPt+NLt)*128, NCt*128);
}
